// NeuralCDE_21852793602429
// MI455X (gfx1250) — matrix operand order repaired and checked by reading, not compiled, not run on hardware
//
#include <hip/hip_runtime.h>
#include <hip/hip_bf16.h>

// Neural CDE (ReversibleHeun scan, T=1024) for MI455X / gfx1250.
// Single persistent 1024-thread workgroup (32 wave32 waves on one WGP):
//  - all per-step GEMMs via v_wmma_f32_16x16x32_f16
//  - small weights pre-packed to B-fragment layout in LDS (fp16)
//  - big cw2 (1088x128) pre-packed fp16 fragments in global scratch, streamed
//    from L2 each step
//  - y / yhat / e_prev state held in registers (8 elements per thread)
//  - activations use single-instruction TRANS ops (v_rcp_f32 / v_tanh_f32)
//    instead of IEEE division chains, so the epilogue co-executes with WMMA.

typedef __attribute__((ext_vector_type(16))) _Float16 v16h;
typedef __attribute__((ext_vector_type(8)))  _Float16 v8h;
typedef __attribute__((ext_vector_type(8)))  float    v8f;

#define B_    128
#define T_    1024
#define C_    17
#define HID_  64
#define WID_  128
#define NOUT3 (HID_ * C_)   // 1088
#define NT3   (NOUT3 / 16)  // 68 N-tiles for the big GEMM

// ---- LDS layout (bytes) -----------------------------------------------------
constexpr int OFF_W0Y  = 0;        // W0y fragments fp16: 2kt*8nt*32*16*2 = 16384
constexpr int OFF_W1   = 16384;    // cw1 fragments fp16: 4*8*32*16*2     = 32768
constexpr int OFF_YH16 = 49152;    // yhat fp16 [128][64]                 = 16384
constexpr int OFF_H1   = 65536;    // h1 fp16 [128][128]                  = 32768
constexpr int OFF_H2   = 98304;    // h2 fp16 [128][128]                  = 32768
constexpr int OFF_SCRA = 131072;   // f32 scratch [128][128]              = 65536
constexpr int OFF_CB0E = 196608;   // effective bias0 [128] f32           = 512
constexpr int OFF_CB1  = 197120;   // cb1 [128] f32                       = 512
constexpr int OFF_CB2  = 197632;   // cb2 [1088] f32                      = 4352
constexpr int OFF_W0T  = 201984;   // cw0[:,0] time column [128] f32      = 512
constexpr int OFF_INC  = 202496;   // inc [128][17] f32                   = 8704
constexpr int SMEM_TOTAL = 211200; // ~206.25 KB (<= 320 KB per WGP)

// ---- Scratch (d_ws) layout --------------------------------------------------
// [0, 557056)            : f buffer, f32 [128][1088]
// [557056, 835584)       : cw2 fragments fp16 [4kt][68nt][32lane][16]
constexpr size_t WS_F_BYTES  = (size_t)B_ * NOUT3 * 4;   // 557056
constexpr size_t WS_W2_OFF_F = WS_F_BYTES / 4;           // float offset

// ---- Fast transcendental helpers (single TRANS instructions) ----------------
__device__ __forceinline__ float fast_rcp(float x) {
#if __has_builtin(__builtin_amdgcn_rcpf)
  return __builtin_amdgcn_rcpf(x);      // v_rcp_f32
#else
  return 1.0f / x;
#endif
}
__device__ __forceinline__ float lipswish_f(float x) {
  // 0.909 * x * sigmoid(x);  sigmoid = rcp(1 + exp(-x))
  return 0.909f * x * fast_rcp(1.0f + __expf(-x));
}
__device__ __forceinline__ float tanh_f(float x) {
#if __has_builtin(__builtin_amdgcn_tanhf)
  return __builtin_amdgcn_tanhf(x);     // v_tanh_f32
#else
  x = fminf(fmaxf(x, -15.0f), 15.0f);
  float e = __expf(2.0f * x);
  return (e - 1.0f) * fast_rcp(e + 1.0f);
#endif
}

// Pre-pack a row-major f32 weight W[Nout][rowStride] (cols colOff..colOff+K)
// into fp16 B-matrix fragments: dst[((kt*NT+tn)*32 + lane)*16 + i] =
//   W[tn*16 + (lane&15)][colOff + kt*32 + 16*(lane>>4) + i]
__device__ __forceinline__ void prepack_bfrag(const float* __restrict__ W,
                                              int rowStride, int colOff,
                                              _Float16* __restrict__ dst,
                                              int NT, int KT, int tid) {
  int total = KT * NT * 32;
  for (int idx = tid; idx < total; idx += 1024) {
    int lane = idx & 31;
    int tnkt = idx >> 5;
    int tn = tnkt % NT, kt = tnkt / NT;
    int j  = tn * 16 + (lane & 15);
    int kb = colOff + kt * 32 + (lane >> 4) * 16;
    const float* src = W + j * rowStride + kb;
    _Float16* d = dst + (size_t)idx * 16;
#pragma unroll
    for (int i = 0; i < 16; ++i) d[i] = (_Float16)src[i];
  }
}

// Tiled GEMM phase: OUT[128][NT*16] = act( ACT[128][K] x Wfrag^T + bias ).
// mode 0: lipswish -> fp16 LDS buffer.  mode 1: tanh -> f32 global buffer.
__device__ __forceinline__ void mm_tiles(const _Float16* __restrict__ act,
                                         int actStride,
                                         const _Float16* __restrict__ wfrag,
                                         const float* __restrict__ bias,
                                         int NT, int KT, int mode,
                                         _Float16* __restrict__ out16,
                                         int outStride,
                                         float* __restrict__ outf,
                                         int outfStride, int wave, int lane) {
  const int half = lane >> 4, ln = lane & 15;
  const int total = 8 * NT;  // 8 M-tiles (B=128)
  for (int tile = wave; tile < total; tile += 32) {
    const int tm = tile & 7, tn = tile >> 3;
    const float bv = bias[tn * 16 + ln];
    v8f acc;
#pragma unroll
    for (int r = 0; r < 8; ++r) acc[r] = bv;
    const int m = tm * 16 + ln;
#pragma unroll 4
    for (int kt = 0; kt < KT; ++kt) {
      // A fragment: two contiguous 16B chunks from row-major LDS activations
      const int kb = kt * 32 + half * 16;
      const v8h lo = *(const v8h*)(act + m * actStride + kb);
      const v8h hi = *(const v8h*)(act + m * actStride + kb + 8);
      v16h a;
#pragma unroll
      for (int i = 0; i < 8; ++i) { a[i] = lo[i]; a[i + 8] = hi[i]; }
      // B fragment: contiguous 32B pre-packed
      const _Float16* bp = wfrag + ((size_t)((kt * NT + tn) * 32 + lane)) * 16;
      if (mode == 1 && kt + 1 < KT)  // stream next fragment from L2
        __builtin_prefetch(wfrag + ((size_t)(((kt + 1) * NT + tn) * 32 + lane)) * 16, 0, 1);
      v16h bfr = *(const v16h*)bp;
      acc = __builtin_amdgcn_wmma_f32_16x16x32_f16(
          false, a, false, bfr, (short)0, acc, false, false);
    }
    const int n = tn * 16 + ln;
#pragma unroll
    for (int r = 0; r < 8; ++r) {
      const int mr = tm * 16 + half * 8 + r;
      const float v = acc[r];
      if (mode == 0)
        out16[mr * outStride + n] = (_Float16)lipswish_f(v);
      else
        outf[(size_t)mr * outfStride + n] = tanh_f(v);
    }
  }
}

extern "C" __global__ void __launch_bounds__(1024, 1)
ncde_persistent_kernel(const float* __restrict__ ts, const float* __restrict__ ys,
                       const float* __restrict__ iw0, const float* __restrict__ ib0,
                       const float* __restrict__ iw1, const float* __restrict__ ib1,
                       const float* __restrict__ iw2, const float* __restrict__ ib2,
                       const float* __restrict__ cw0, const float* __restrict__ cb0,
                       const float* __restrict__ cw1, const float* __restrict__ cb1,
                       const float* __restrict__ cw2, const float* __restrict__ cb2,
                       const float* __restrict__ rw,  const float* __restrict__ rb,
                       float* __restrict__ out, float* __restrict__ ws) {
  extern __shared__ char smem[];
  _Float16* w0yF   = (_Float16*)(smem + OFF_W0Y);
  _Float16* w1F    = (_Float16*)(smem + OFF_W1);
  _Float16* yhat16 = (_Float16*)(smem + OFF_YH16);
  _Float16* h1buf  = (_Float16*)(smem + OFF_H1);
  _Float16* h2buf  = (_Float16*)(smem + OFF_H2);
  float*    scrA   = (float*)(smem + OFF_SCRA);
  float*    cb0eff = (float*)(smem + OFF_CB0E);
  float*    cb1l   = (float*)(smem + OFF_CB1);
  float*    cb2l   = (float*)(smem + OFF_CB2);
  float*    w0t    = (float*)(smem + OFF_W0T);
  float*    incl   = (float*)(smem + OFF_INC);

  float*    fbuf = ws;                              // [128][1088] f32
  _Float16* w2F  = (_Float16*)(ws + WS_W2_OFF_F);   // cw2 fragments fp16

  const int tid = threadIdx.x, lane = tid & 31, wave = tid >> 5;
  const int b = tid >> 3;            // batch row owned by this thread
  const int hb = (tid & 7) * 8;      // hidden slice [hb, hb+8)

  // ---- Phase 0: pre-pack weights + load biases ------------------------------
  prepack_bfrag(cw0, HID_ + 1, 1, w0yF, 8, 2, tid);   // W0y = cw0[:,1:65]
  prepack_bfrag(cw1, WID_, 0, w1F, 8, 4, tid);        // cw1 [128][128]
  prepack_bfrag(cw2, WID_, 0, w2F, NT3, 4, tid);      // cw2 [1088][128] -> ws
  for (int i = tid; i < WID_; i += 1024) {
    cb1l[i] = cb1[i];
    w0t[i]  = cw0[i * (HID_ + 1)];   // time column
  }
  for (int i = tid; i < NOUT3; i += 1024) cb2l[i] = cb2[i];
  __threadfence();                   // publish w2F fragments
  __syncthreads();

  // ---- Initial MLP (one-time, scalar VALU) ----------------------------------
  {  // h0 = relu(X0 @ iw0.T + ib0) -> scrA
    const int jb = (tid & 7) * 16;
    float x0[C_];
    x0[0] = ts[0];
#pragma unroll
    for (int c = 0; c < 16; ++c) x0[c + 1] = ys[(size_t)b * T_ * 16 + c];
    for (int j = jb; j < jb + 16; ++j) {
      float s = ib0[j];
#pragma unroll
      for (int c = 0; c < C_; ++c) s += x0[c] * iw0[j * C_ + c];
      scrA[b * WID_ + j] = fmaxf(s, 0.0f);
    }
  }
  __syncthreads();
  {  // h1i = relu(h0 @ iw1.T + ib1) -> scrA (in-place via registers)
    const int jb = (tid & 7) * 16;
    float r[16];
    for (int jj = 0; jj < 16; ++jj) {
      const int j = jb + jj;
      float s = ib1[j];
      for (int k = 0; k < WID_; ++k) s += scrA[b * WID_ + k] * iw1[j * WID_ + k];
      r[jj] = fmaxf(s, 0.0f);
    }
    __syncthreads();
    for (int jj = 0; jj < 16; ++jj) scrA[b * WID_ + jb + jj] = r[jj];
  }
  __syncthreads();

  float y_r[8], yh_r[8], ep_r[8];
  for (int hh = 0; hh < 8; ++hh) {   // y0 = h1i @ iw2.T + ib2 (state registers)
    const int h = hb + hh;
    float s = ib2[h];
    for (int k = 0; k < WID_; ++k) s += scrA[b * WID_ + k] * iw2[h * WID_ + k];
    y_r[hh] = s;
    yh_r[hh] = s;
    yhat16[b * HID_ + h] = (_Float16)s;
  }
  if (tid < WID_) cb0eff[tid] = cb0[tid] + ts[0] * w0t[tid];
  __syncthreads();

  // ---- f0 = vf(ts[0], y0) ---------------------------------------------------
  mm_tiles(yhat16, HID_, w0yF, cb0eff, 8, 2, 0, h1buf, WID_, nullptr, 0, wave, lane);
  __syncthreads();
  mm_tiles(h1buf, WID_, w1F, cb1l, 8, 4, 0, h2buf, WID_, nullptr, 0, wave, lane);
  __syncthreads();
  mm_tiles(h2buf, WID_, w2F, cb2l, NT3, 4, 1, nullptr, 0, fbuf, NOUT3, wave, lane);
  __threadfence();
  __syncthreads();

  // ---- ReversibleHeun scan: 1023 sequential steps ---------------------------
  for (int n = 0; n < T_ - 1; ++n) {
    const float t1 = ts[n + 1];
    // Phase 1: control increments + effective bias0(t1)
    for (int idx = tid; idx < B_ * C_; idx += 1024) {
      const int bb = idx / C_, c = idx % C_;
      float v;
      if (c == 0) v = t1 - ts[n];
      else {
        const size_t base = (size_t)bb * T_ * 16 + (c - 1);
        v = ys[base + (size_t)(n + 1) * 16] - ys[base + (size_t)n * 16];
      }
      incl[bb * C_ + c] = v;
    }
    if (tid < WID_) cb0eff[tid] = cb0[tid] + t1 * w0t[tid];
    __syncthreads();

    // Phase 2: e_prev = f_prev . inc ; yhat1 = 2y - yhat + e_prev
    for (int hh = 0; hh < 8; ++hh) {
      const int h = hb + hh;
      const float* fp = fbuf + (size_t)b * NOUT3 + h * C_;
      float e = 0.0f;
#pragma unroll
      for (int c = 0; c < C_; ++c) e += fp[c] * incl[b * C_ + c];
      ep_r[hh] = e;
      const float yh1 = 2.0f * y_r[hh] - yh_r[hh] + e;
      yh_r[hh] = yh1;
      yhat16[b * HID_ + h] = (_Float16)yh1;
    }
    __syncthreads();

    // Phase 3-5: f1 = vf(t1, yhat1) via three WMMA GEMMs
    mm_tiles(yhat16, HID_, w0yF, cb0eff, 8, 2, 0, h1buf, WID_, nullptr, 0, wave, lane);
    __syncthreads();
    mm_tiles(h1buf, WID_, w1F, cb1l, 8, 4, 0, h2buf, WID_, nullptr, 0, wave, lane);
    __syncthreads();
    mm_tiles(h2buf, WID_, w2F, cb2l, NT3, 4, 1, nullptr, 0, fbuf, NOUT3, wave, lane);
    __threadfence();
    __syncthreads();

    // Phase 6: e_new = f1 . inc ; y1 = y + 0.5*(e_prev + e_new)
    for (int hh = 0; hh < 8; ++hh) {
      const int h = hb + hh;
      const float* fp = fbuf + (size_t)b * NOUT3 + h * C_;
      float e = 0.0f;
#pragma unroll
      for (int c = 0; c < C_; ++c) e += fp[c] * incl[b * C_ + c];
      y_r[hh] += 0.5f * (ep_r[hh] + e);
    }
    __syncthreads();  // protect incl / fbuf for next iteration
  }

  // ---- Readout: out[b] = rb + sum_h y[b][h]*rw[h] ---------------------------
  float part = 0.0f;
#pragma unroll
  for (int hh = 0; hh < 8; ++hh) part += y_r[hh] * rw[hb + hh];
  scrA[b * 8 + (tid & 7)] = part;
  __syncthreads();
  if (tid < B_) {
    float s = rb[0];
#pragma unroll
    for (int g = 0; g < 8; ++g) s += scrA[tid * 8 + g];
    out[tid] = s;
  }
}

extern "C" void kernel_launch(void* const* d_in, const int* in_sizes, int n_in,
                              void* d_out, int out_size, void* d_ws, size_t ws_size,
                              hipStream_t stream) {
  const float* ts  = (const float*)d_in[0];
  const float* ys  = (const float*)d_in[1];
  const float* iw0 = (const float*)d_in[2];
  const float* ib0 = (const float*)d_in[3];
  const float* iw1 = (const float*)d_in[4];
  const float* ib1 = (const float*)d_in[5];
  const float* iw2 = (const float*)d_in[6];
  const float* ib2 = (const float*)d_in[7];
  const float* cw0 = (const float*)d_in[8];
  const float* cb0 = (const float*)d_in[9];
  const float* cw1 = (const float*)d_in[10];
  const float* cb1 = (const float*)d_in[11];
  const float* cw2 = (const float*)d_in[12];
  const float* cb2 = (const float*)d_in[13];
  const float* rw  = (const float*)d_in[14];
  const float* rb  = (const float*)d_in[15];

  (void)in_sizes; (void)n_in; (void)out_size; (void)ws_size;

  hipFuncSetAttribute((const void*)ncde_persistent_kernel,
                      hipFuncAttributeMaxDynamicSharedMemorySize, SMEM_TOTAL);

  ncde_persistent_kernel<<<1, 1024, SMEM_TOTAL, stream>>>(
      ts, ys, iw0, ib0, iw1, ib1, iw2, ib2,
      cw0, cb0, cw1, cb1, cw2, cb2, rw, rb,
      (float*)d_out, (float*)d_ws);
}
